// MultiheadAttentionBlock_56564719289046
// MI455X (gfx1250) — matrix operand order repaired and checked by reading, not compiled, not run on hardware
//
#include <hip/hip_runtime.h>
#include <stdint.h>

typedef __attribute__((ext_vector_type(16))) __bf16 v16bf;
typedef __attribute__((ext_vector_type(8)))  float  v8f;

#define BB 2
#define HH 16
#define DD 128
#define SS 2048
#define TT 4096
#define EE 2048

static __device__ __forceinline__ unsigned short f2bf(float f) {
    union { float f; unsigned int u; } c; c.f = f;
    unsigned int u = c.u;
    u += 0x7FFFu + ((u >> 16) & 1u);          // round-to-nearest-even
    return (unsigned short)(u >> 16);
}

union Frag16 { uint4 u[2]; v16bf v; };

static __device__ __forceinline__ v8f v8f_zero() {
    v8f z = {0.f, 0.f, 0.f, 0.f, 0.f, 0.f, 0.f, 0.f};
    return z;
}

// async copy of 16 bytes global -> LDS, tracked by ASYNCcnt (CDNA5 path)
static __device__ __forceinline__ void async_b128(const void* lds_dst, const void* gsrc) {
    unsigned la = (unsigned)(uintptr_t)lds_dst;            // LDS byte offset (addr[31:0])
    unsigned long long ga = (unsigned long long)(uintptr_t)gsrc;
    asm volatile("global_load_async_to_lds_b128 %0, %1, off"
                 :: "v"(la), "v"(ga) : "memory");
}
static __device__ __forceinline__ void wait_async0() {
    asm volatile("s_wait_asynccnt 0x0" ::: "memory");
}

// ---------------------------------------------------------------- converts
__global__ void convert_f32_bf16(const float* __restrict__ src,
                                 unsigned short* __restrict__ dst, int n) {
    int i = blockIdx.x * 256 + threadIdx.x;
    if (i < n) dst[i] = f2bf(src[i]);
}

// copy caches into concat K/V outputs (f32) + bf16 mirror for attention
__global__ void cache_concat(const float* __restrict__ kc, const float* __restrict__ vc,
                             float* __restrict__ Kf, float* __restrict__ Vf,
                             unsigned short* __restrict__ Kb, unsigned short* __restrict__ Vb) {
    int i = blockIdx.x * 256 + threadIdx.x;         // over B*H*2048*128
    const int n = BB * HH * SS * DD;
    if (i >= n) return;
    int bh = i / (SS * DD);
    int r  = i - bh * (SS * DD);
    int o  = bh * (TT * DD) + r;                    // rows [0,2048) of concat
    float k = kc[i], v = vc[i];
    Kf[o] = k; Vf[o] = v;
    Kb[o] = f2bf(k); Vb[o] = f2bf(v);
}

// ---------------------------------------------------------------- GEMM
// y[m,n] = sum_k A[m,k] * B[n,k]   (torch Linear: x @ W^T), bf16 in, f32 acc
// Double-buffered LDS tiles fed by global_load_async_to_lds_b128.
// mode 0: out is [M,N] row-major. mode 1: KV scatter to [b,h,2048+s,d].
__global__ __launch_bounds__(256)
void gemm_bf16(const unsigned short* __restrict__ A,
               const unsigned short* __restrict__ Bm,
               int M, int N, int K,
               float* __restrict__ outF, unsigned short* __restrict__ outH, int mode) {
    __shared__ __align__(16) unsigned short As[2][128][32];
    __shared__ __align__(16) unsigned short Bs[2][128][32];

    const int tid  = threadIdx.x;
    const int wave = tid >> 5, lane = tid & 31;
    const int laneN = lane & 15, hi = lane >> 4;
    const int m0 = blockIdx.y * 128, n0 = blockIdx.x * 128;
    const int wm = wave & 1;        // 2 waves along M (64 rows each)
    const int wn = wave >> 1;       // 4 waves along N (32 cols each)

    v8f acc[4][2];
    for (int i = 0; i < 4; ++i)
        for (int j = 0; j < 2; ++j) acc[i][j] = v8f_zero();

    // per-thread staging coordinates: 2x uint4 per tile per operand
    const int r0 = (tid * 2) >> 2;           // rows for idx = tid*2, tid*2+1
    const int c0 = ((tid * 2) & 3) * 8;

    auto stage = [&](int buf, int kt) {
        for (int it = 0; it < 2; ++it) {
            int idx = tid + it * 256;        // 512 uint4 per tile
            int row = idx >> 2;
            int c   = (idx & 3) * 8;         // in halves
            async_b128(&As[buf][row][c], A  + (size_t)(m0 + row) * K + kt + c);
            async_b128(&Bs[buf][row][c], Bm + (size_t)(n0 + row) * K + kt + c);
        }
        // speculative prefetch one more tile ahead into L2
        if (kt + 32 < K) {
            __builtin_prefetch(A  + (size_t)(m0 + r0) * K + kt + 32 + c0, 0, 1);
            __builtin_prefetch(Bm + (size_t)(n0 + r0) * K + kt + 32 + c0, 0, 1);
        }
    };

    stage(0, 0);
    int pb = 0;
    for (int kt = 0; kt < K; kt += 32, pb ^= 1) {
        wait_async0();                        // this wave's async tile writes done
        __syncthreads();                      // all waves' tiles visible
        if (kt + 32 < K) stage(pb ^ 1, kt + 32);

        // B fragments: lane = N, K chunks {kb..kb+7, kb+16..kb+23}, kb = hi*8
        Frag16 b0, b1;
        {
            int n  = wn * 32 + laneN;
            int kb = hi * 8;
            b0.u[0] = *(const uint4*)&Bs[pb][n][kb];
            b0.u[1] = *(const uint4*)&Bs[pb][n][kb + 16];
            b1.u[0] = *(const uint4*)&Bs[pb][n + 16][kb];
            b1.u[1] = *(const uint4*)&Bs[pb][n + 16][kb + 16];
        }
        for (int i = 0; i < 4; ++i) {
            // A fragment: lane = M, K chunks {kb..kb+7, kb+16..kb+23}, kb = hi*8
            Frag16 a;
            int mrow = wm * 64 + i * 16 + laneN;
            int kb   = hi * 8;
            a.u[0] = *(const uint4*)&As[pb][mrow][kb];
            a.u[1] = *(const uint4*)&As[pb][mrow][kb + 16];
            acc[i][0] = __builtin_amdgcn_wmma_f32_16x16x32_bf16(
                false, a.v, false, b0.v, (short)0, acc[i][0], false, false);
            acc[i][1] = __builtin_amdgcn_wmma_f32_16x16x32_bf16(
                false, a.v, false, b1.v, (short)0, acc[i][1], false, false);
        }
    }

    for (int i = 0; i < 4; ++i)
        for (int j = 0; j < 2; ++j)
            for (int v = 0; v < 8; ++v) {
                int m = m0 + wm * 64 + i * 16 + v + hi * 8;
                int n = n0 + wn * 32 + j * 16 + laneN;
                float val = acc[i][j][v];
                if (mode == 0) {
                    size_t o = (size_t)m * N + n;
                    if (outF) outF[o] = val;
                    if (outH) outH[o] = f2bf(val);
                } else {
                    int b = m >> 11, s = m & 2047;
                    int h = n >> 7,  d = n & 127;
                    size_t o = ((size_t)(b * HH + h) * TT + SS + s) * DD + d;
                    if (outF) outF[o] = val;
                    if (outH) outH[o] = f2bf(val);
                }
            }
}

// ---------------------------------------------------------------- attention
// flash-style: Q tile 64 rows, K/V tiles 32 keys, online softmax, all WMMA bf16
__global__ __launch_bounds__(256)
void attention_fwd(const unsigned short* __restrict__ Qb,
                   const unsigned short* __restrict__ Kb,
                   const unsigned short* __restrict__ Vb,
                   const float* __restrict__ mask,
                   unsigned short* __restrict__ AOb) {
    __shared__ __align__(16) unsigned short Qs[64][128];   // 16 KB
    __shared__ __align__(16) unsigned short Ks[32][128];   //  8 KB
    __shared__ __align__(16) unsigned short Vt[128][32];   //  8 KB (transposed)
    __shared__ __align__(16) float          Ssm[64][32];   //  8 KB
    __shared__ __align__(16) unsigned short Ps[64][32];    //  4 KB
    __shared__ float row_m[64], row_l[64], row_alpha[64];

    const int tid  = threadIdx.x;
    const int wave = tid >> 5, lane = tid & 31;
    const int laneN = lane & 15, hi = lane >> 4;
    const int qt = blockIdx.x, h = blockIdx.y, b = blockIdx.z;
    const int qbase = qt * 64;
    const float scale = 0.08838834764831845f;   // 1/sqrt(128)

    // async-stage Q tile (64 x 128 bf16)
    for (int it = 0; it < 4; ++it) {
        int idx = tid + it * 256;                // 1024 uint4
        int r = idx >> 4, c = (idx & 15) * 8;
        async_b128(&Qs[r][c], Qb + (size_t)(b * SS + qbase + r) * EE + h * DD + c);
    }
    if (tid < 64) { row_m[tid] = -3.0e38f; row_l[tid] = 0.f; }

    v8f o[4];
    for (int j = 0; j < 4; ++j) o[j] = v8f_zero();

    const size_t kvbase = (size_t)(b * HH + h) * TT * DD;
    const int mi = wave & 3;                     // M tile (rows mi*16..)
    const int nt = wave >> 2;                    // score N tile for QK^T (0..1)
    const int jb = (wave >> 2) * 4;              // O tile group (d-tiles)

    for (int t0 = 0; t0 < TT; t0 += 32) {
        // async K (row-major); V staged transposed with scalar ds stores
        for (int it = 0; it < 2; ++it) {
            int idx = tid + it * 256;            // 512 uint4
            int r = idx >> 4, c = (idx & 15) * 8;
            async_b128(&Ks[r][c], Kb + kvbase + (size_t)(t0 + r) * DD + c);
        }
        for (int it = 0; it < 16; ++it) {
            int idx = tid + it * 256;            // 4096 halves
            int r = idx >> 7, d = idx & 127;
            Vt[d][r] = Vb[kvbase + (size_t)(t0 + r) * DD + d];
        }
        wait_async0();                           // covers Q on first pass + K tile
        __syncthreads();

        // S = Q K^T  (each wave: one 16x16 score tile, k-loop over D=128)
        {
            v8f sacc = v8f_zero();
            for (int kk = 0; kk < 4; ++kk) {
                Frag16 a;
                int mrow = mi * 16 + laneN;
                int kb = kk * 32 + hi * 8;
                a.u[0] = *(const uint4*)&Qs[mrow][kb];
                a.u[1] = *(const uint4*)&Qs[mrow][kb + 16];
                Frag16 bk;
                int trow = nt * 16 + laneN;
                int kb2 = kk * 32 + hi * 8;
                bk.u[0] = *(const uint4*)&Ks[trow][kb2];
                bk.u[1] = *(const uint4*)&Ks[trow][kb2 + 16];
                sacc = __builtin_amdgcn_wmma_f32_16x16x32_bf16(
                    false, a.v, false, bk.v, (short)0, sacc, false, false);
            }
            for (int v = 0; v < 8; ++v)
                Ssm[mi * 16 + v + hi * 8][nt * 16 + laneN] = sacc[v];
        }
        __syncthreads();

        // online softmax over the 64x32 score tile
        {
            const int r  = tid >> 2;             // row 0..63
            const int cg = (tid & 3) * 8;        // 8 cols per thread
            float m_old = row_m[r];
            float lmax = -3.0e38f;
            for (int c = 0; c < 8; ++c) {
                float s = Ssm[r][cg + c] * scale +
                          mask[(size_t)(qbase + r) * TT + t0 + cg + c];
                Ssm[r][cg + c] = s;
                lmax = fmaxf(lmax, s);
            }
            lmax = fmaxf(lmax, __shfl_xor(lmax, 1, 32));
            lmax = fmaxf(lmax, __shfl_xor(lmax, 2, 32));
            float m_new = fmaxf(m_old, lmax);
            float lsum = 0.f;
            for (int c = 0; c < 8; ++c) {
                float p = __expf(Ssm[r][cg + c] - m_new);
                Ps[r][cg + c] = f2bf(p);
                lsum += p;
            }
            lsum += __shfl_xor(lsum, 1, 32);
            lsum += __shfl_xor(lsum, 2, 32);
            if ((tid & 3) == 0) {
                float alpha = __expf(m_old - m_new);
                row_alpha[r] = alpha;
                row_l[r] = row_l[r] * alpha + lsum;
                row_m[r] = m_new;
            }
        }
        __syncthreads();

        // rescale O and accumulate P @ V  (each wave: 4 O tiles of 16x16)
        {
            for (int j = 0; j < 4; ++j)
                for (int v = 0; v < 8; ++v)
                    o[j][v] *= row_alpha[mi * 16 + v + hi * 8];
            Frag16 a;
            int mrow = mi * 16 + laneN;
            int kb = hi * 8;
            a.u[0] = *(const uint4*)&Ps[mrow][kb];
            a.u[1] = *(const uint4*)&Ps[mrow][kb + 16];
            for (int j = 0; j < 4; ++j) {
                Frag16 bv;
                int drow = (jb + j) * 16 + laneN;
                int kb2 = hi * 8;
                bv.u[0] = *(const uint4*)&Vt[drow][kb2];
                bv.u[1] = *(const uint4*)&Vt[drow][kb2 + 16];
                o[j] = __builtin_amdgcn_wmma_f32_16x16x32_bf16(
                    false, a.v, false, bv.v, (short)0, o[j], false, false);
            }
        }
        __syncthreads();
    }

    // normalize and emit bf16 attention output [b, s, h*128 + d]
    for (int j = 0; j < 4; ++j)
        for (int v = 0; v < 8; ++v) {
            int mr = mi * 16 + v + hi * 8;
            int d  = (jb + j) * 16 + laneN;
            float val = o[j][v] / row_l[mr];
            AOb[(size_t)(b * SS + qbase + mr) * EE + h * DD + d] = f2bf(val);
        }
}

// ---------------------------------------------------------------- launch
extern "C" void kernel_launch(void* const* d_in, const int* in_sizes, int n_in,
                              void* d_out, int out_size, void* d_ws, size_t ws_size,
                              hipStream_t stream) {
    const float* x    = (const float*)d_in[0];
    const float* Wq   = (const float*)d_in[1];
    const float* Wk   = (const float*)d_in[2];
    const float* Wv   = (const float*)d_in[3];
    const float* Wo   = (const float*)d_in[4];
    const float* mask = (const float*)d_in[5];
    const float* kc   = (const float*)d_in[6];
    const float* vc   = (const float*)d_in[7];

    float* outO = (float*)d_out;                          // [2,2048,2048]
    float* outK = outO + (size_t)BB * SS * EE;            // [2,16,4096,128]
    float* outV = outK + (size_t)BB * HH * TT * DD;       // [2,16,4096,128]

    size_t off = 0;
    auto wsAlloc = [&](size_t bytes) -> void* {
        void* p = (char*)d_ws + off;
        off += (bytes + 255) & ~(size_t)255;
        return p;
    };
    unsigned short* xb  = (unsigned short*)wsAlloc((size_t)BB * SS * EE * 2);
    unsigned short* Wqb = (unsigned short*)wsAlloc((size_t)EE * EE * 2);
    unsigned short* Wkb = (unsigned short*)wsAlloc((size_t)EE * EE * 2);
    unsigned short* Wvb = (unsigned short*)wsAlloc((size_t)EE * EE * 2);
    unsigned short* Wob = (unsigned short*)wsAlloc((size_t)EE * EE * 2);
    unsigned short* Qb  = (unsigned short*)wsAlloc((size_t)BB * SS * EE * 2);
    unsigned short* Kb  = (unsigned short*)wsAlloc((size_t)BB * HH * TT * DD * 2);
    unsigned short* Vb  = (unsigned short*)wsAlloc((size_t)BB * HH * TT * DD * 2);
    unsigned short* AOb = (unsigned short*)wsAlloc((size_t)BB * SS * EE * 2);

    const int nx = BB * SS * EE;      // 8388608
    const int nw = EE * EE;           // 4194304
    convert_f32_bf16<<<nx / 256, 256, 0, stream>>>(x,  xb,  nx);
    convert_f32_bf16<<<nw / 256, 256, 0, stream>>>(Wq, Wqb, nw);
    convert_f32_bf16<<<nw / 256, 256, 0, stream>>>(Wk, Wkb, nw);
    convert_f32_bf16<<<nw / 256, 256, 0, stream>>>(Wv, Wvb, nw);
    convert_f32_bf16<<<nw / 256, 256, 0, stream>>>(Wo, Wob, nw);

    const int nc = BB * HH * SS * DD; // 8388608
    cache_concat<<<nc / 256, 256, 0, stream>>>(kc, vc, outK, outV, Kb, Vb);

    dim3 gg(EE / 128, (BB * SS) / 128);   // (16, 32)
    gemm_bf16<<<gg, 256, 0, stream>>>(xb, Wqb, BB * SS, EE, EE, nullptr, Qb, 0);
    gemm_bf16<<<gg, 256, 0, stream>>>(xb, Wkb, BB * SS, EE, EE, outK, Kb, 1);
    gemm_bf16<<<gg, 256, 0, stream>>>(xb, Wvb, BB * SS, EE, EE, outV, Vb, 1);

    dim3 ga(SS / 64, HH, BB);             // (32, 16, 2)
    attention_fwd<<<ga, 256, 0, stream>>>(Qb, Kb, Vb, mask, AOb);

    gemm_bf16<<<gg, 256, 0, stream>>>(AOb, Wob, BB * SS, EE, EE, outO, nullptr, 0);
}
